// rnn_new_73212012528064
// MI455X (gfx1250) — matrix operand order repaired and checked by reading, not compiled, not run on hardware
//
#include <hip/hip_runtime.h>
#include <hip/hip_bf16.h>
#include <math.h>

// CDNA5 / gfx1250: wave32, WMMA 16x16x32 bf16 -> f32
typedef __attribute__((ext_vector_type(16))) __bf16       v16bf;
typedef __attribute__((ext_vector_type(8)))  float        v8f;
typedef __attribute__((ext_vector_type(4)))  unsigned int v4u;

#define TT        31     // tree slots / decode steps
#define BINC      4      // binary code length
#define FEA       9
#define HID       128
#define VOCAB     19
#define NC        40
#define INSZ      133    // TT*BINC + FEA
#define KZ        261    // INSZ + HID
#define KPAD      288    // 9 * 32
#define NKCH      9      // K chunks of 32 for main GEMM
#define NOUT      59     // VOCAB + NC (head GEMM N, padded to 64)
#define ROWS      16     // batch rows per block == WMMA M
#define CONST_ID  18
#define LEAF_START 6

union AFrag { v16bf v; v4u q[2]; };

__global__ __launch_bounds__(256, 1) void tree_rnn_wmma(
    const float* __restrict__ x,
    const float* __restrict__ ntok,
    const float* __restrict__ ncon,
    const float* __restrict__ Wih,
    const float* __restrict__ Whh,
    const float* __restrict__ bih,
    const float* __restrict__ bhh,
    const float* __restrict__ Wout,
    const float* __restrict__ bout,
    const float* __restrict__ Wc,
    const float* __restrict__ bc,
    int*   __restrict__ seq_out,
    float* __restrict__ cval_out,
    float* __restrict__ lp_out,
    int BS)
{
  __shared__ __align__(16) __bf16 z[ROWS][KPAD];    // [xin | h] bf16, zero-padded tail
  __shared__ __align__(16) __bf16 hnb[ROWS][HID];   // h_new bf16 for head GEMM A
  __shared__ float lg[ROWS][64];                    // fused logits: [0,19)=tok, [19,59)=const
  __shared__ float ntk_s[2][ROWS][VOCAB];           // double-buffered coalesced noise staging
  __shared__ float nct_s[2][ROWS][NC];
  __shared__ int   st_stack[ROWS][TT];
  __shared__ int   st_pos[ROWS];
  __shared__ int   st_sp[ROWS];
  __shared__ int   st_act[ROWS];
  __shared__ float st_lp[ROWS];
  __shared__ int   st_choice[ROWS];
  __shared__ float st_lpch[ROWS];
  __shared__ int   st_cch[ROWS];
  __shared__ float st_lpc[ROWS];

  const int tid  = threadIdx.x;
  const int lane = tid & 31;
  const int wave = tid >> 5;
  const int lo16 = lane & 15;
  const int half = lane >> 4;
  const int row0 = blockIdx.x * ROWS;

  // ---------------- init: z = [0^(124) | x | h=0 | pad], per-row decoder state ----------------
  for (int i = tid; i < ROWS * KPAD; i += 256) {
    int m = i / KPAD, k = i - m * KPAD;
    float v = 0.0f;
    if (k >= TT * BINC && k < INSZ) v = x[(size_t)(row0 + m) * FEA + (k - TT * BINC)];
    z[m][k] = (__bf16)v;
  }
  if (tid < ROWS) { st_pos[tid] = 0; st_sp[tid] = 0; st_act[tid] = 1; st_lp[tid] = 0.0f; }
  for (int i = tid; i < ROWS * TT; i += 256) {
    st_stack[i / TT][i - (i / TT) * TT] = 0;
    seq_out[(size_t)row0 * TT + i]  = -1;     // rows consecutive => flat init
    cval_out[(size_t)row0 * TT + i] = 0.0f;
  }
  // prefetch step-0 noise (contiguous runs), parity 0
  for (int i = tid; i < ROWS * VOCAB; i += 256)
    ntk_s[0][i / VOCAB][i - (i / VOCAB) * VOCAB] = ntok[(size_t)row0 * VOCAB + i];
  for (int i = tid; i < ROWS * NC; i += 256)
    nct_s[0][i / NC][i - (i / NC) * NC] = ncon[(size_t)row0 * NC + i];

  // ---------------- preload weights into register-resident B fragments ----------------
  // B layout (16-bit, 32x16): lane half h holds K=[8h..8h+7],[16+8h..16+8h+7]
  // (matching A); column n = lane%16 within the tile.
  v16bf Bm[NKCH];
  {
    const int n = wave * 16 + lo16;           // HID column owned by this wave/lane
    for (int c = 0; c < NKCH; ++c) {
      v16bf b;
      #pragma unroll
      for (int e = 0; e < 16; ++e) {
        int k = 32 * c + 8 * half + e + (e >= 8 ? 8 : 0);
        float w = 0.0f;
        if (k < INSZ)    w = Wih[(size_t)n * INSZ + k];
        else if (k < KZ) w = Whh[(size_t)n * HID + (k - INSZ)];
        b[e] = (__bf16)w;
      }
      Bm[c] = b;
    }
  }
  v16bf Bh[4];
  float bias2 = 0.0f;
  if (wave < 4) {
    const int n2 = wave * 16 + lo16;          // fused head column (0..63, valid < 59)
    for (int c = 0; c < 4; ++c) {
      v16bf b;
      #pragma unroll
      for (int e = 0; e < 16; ++e) {
        int k = 32 * c + 8 * half + e + (e >= 8 ? 8 : 0);
        float w = 0.0f;
        if (n2 < VOCAB)      w = Wout[(size_t)n2 * HID + k];
        else if (n2 < NOUT)  w = Wc[(size_t)(n2 - VOCAB) * HID + k];
        b[e] = (__bf16)w;
      }
      Bh[c] = b;
    }
    bias2 = (n2 < VOCAB) ? bout[n2] : ((n2 < NOUT) ? bc[n2 - VOCAB] : 0.0f);
  }
  const float bias1 = bih[wave * 16 + lo16] + bhh[wave * 16 + lo16];

  // ---------------- 31 fused decode steps ----------------
  for (int t = 0; t < TT; ++t) {
    __syncthreads();   // z + state + staged noise from previous step visible

    // GEMM1: h_new_pre(16x128) = z(16x288) @ Wcat^T ; each wave owns 16 columns.
    // Hoist ALL A-fragment LDS loads before the WMMA chain so the DS pipe runs ahead.
    // A layout (16-bit, 16x32): lane row m=lane%16; element runs K=[8h..8h+7],[16+8h..+7].
    AFrag af[NKCH];
    #pragma unroll
    for (int c = 0; c < NKCH; ++c) {
      const __bf16* zp = &z[lo16][32 * c + 8 * half];
      af[c].q[0] = *(const v4u*)zp;
      af[c].q[1] = *(const v4u*)(zp + 16);
    }
    v8f acc;
    #pragma unroll
    for (int r = 0; r < 8; ++r) acc[r] = bias1;
    #pragma unroll
    for (int c = 0; c < NKCH; ++c)
      acc = __builtin_amdgcn_wmma_f32_16x16x32_bf16(false, af[c].v, false, Bm[c],
                                                    (short)0, acc, false, false);
    __syncthreads();   // everyone finished reading z before h is overwritten

    // tanh in-register; C layout: lane col n=lane%16, rows M=r+8*half
    {
      const int n = wave * 16 + lo16;
      #pragma unroll
      for (int r = 0; r < 8; ++r) {
        float hv = tanhf(acc[r]);
        int mm = r + 8 * half;
        __bf16 hb = (__bf16)hv;
        hnb[mm][n] = hb;                       // head GEMM input (all rows)
        if (st_act[mm]) z[mm][INSZ + n] = hb;  // h state update only for active rows
      }
    }
    __syncthreads();   // hnb ready

    if (wave < 4) {
      // GEMM2: fused heads (59x128), waves 0..3
      AFrag ah[4];
      #pragma unroll
      for (int c = 0; c < 4; ++c) {
        const __bf16* hp = &hnb[lo16][32 * c + 8 * half];
        ah[c].q[0] = *(const v4u*)hp;
        ah[c].q[1] = *(const v4u*)(hp + 16);
      }
      v8f acc2;
      #pragma unroll
      for (int r = 0; r < 8; ++r) acc2[r] = bias2;
      #pragma unroll
      for (int c = 0; c < 4; ++c)
        acc2 = __builtin_amdgcn_wmma_f32_16x16x32_bf16(false, ah[c].v, false, Bh[c],
                                                       (short)0, acc2, false, false);
      const int n2 = wave * 16 + lo16;
      #pragma unroll
      for (int r = 0; r < 8; ++r) lg[r + 8 * half][n2] = acc2[r];
    } else {
      // waves 4..7: coalesced prefetch of step t+1 noise into the other parity
      const int tn = t + 1;
      if (tn < TT) {
        const int par = tn & 1;
        for (int i = tid - 128; i < ROWS * VOCAB; i += 128)
          ntk_s[par][i / VOCAB][i - (i / VOCAB) * VOCAB] =
              ntok[((size_t)tn * BS + row0) * VOCAB + i];
        for (int i = tid - 128; i < ROWS * NC; i += 128)
          nct_s[par][i / NC][i - (i / NC) * NC] =
              ncon[((size_t)tn * BS + row0) * NC + i];
      }
    }
    __syncthreads();   // logits ready

    // -------- sampling phase 1: token head on lanes 0-15, const head on lanes 16-31 --------
    const int par = t & 1;
    if (wave == 0) {
      const int rr = lo16;
      if (st_act[rr]) {
        if (half == 0) {
          int p = st_pos[rr];
          p = p < 0 ? 0 : (p > TT - 1 ? TT - 1 : p);
          const int jstart = (p >= 15) ? LEAF_START : 0;   // depth >= MAX_LAYER-1
          float mx = -3.0e38f;
          for (int j = jstart; j < VOCAB; ++j) mx = fmaxf(mx, lg[rr][j]);
          float se = 0.0f;
          for (int j = jstart; j < VOCAB; ++j) se += expf(lg[rr][j] - mx);
          const float lse = mx + logf(se);
          float best = -3.0e38f, lp_ch = 0.0f;
          int choice = jstart;
          for (int j = jstart; j < VOCAB; ++j) {
            float g = -logf(-logf(ntk_s[par][rr][j] + 1e-9f) + 1e-9f);
            float lpj = lg[rr][j] - lse;
            float v = lpj + g;
            if (v > best) { best = v; choice = j; lp_ch = lpj; }
          }
          st_choice[rr] = choice;
          st_lpch[rr]   = lp_ch;
        } else {
          float mxc = -3.0e38f;
          for (int j = 0; j < NC; ++j) mxc = fmaxf(mxc, lg[rr][VOCAB + j]);
          float sec = 0.0f;
          for (int j = 0; j < NC; ++j) sec += expf(lg[rr][VOCAB + j] - mxc);
          const float lsec = mxc + logf(sec);
          float bestc = -3.0e38f, lp_c = 0.0f;
          int cch = 0;
          for (int j = 0; j < NC; ++j) {
            float g = -logf(-logf(nct_s[par][rr][j] + 1e-9f) + 1e-9f);
            float lpj = lg[rr][VOCAB + j] - lsec;
            float v = lpj + g;
            if (v > bestc) { bestc = v; cch = j; lp_c = lpj; }
          }
          st_cch[rr] = cch;
          st_lpc[rr] = lp_c;
        }
      }
    }
    __syncthreads();   // head results visible

    // -------- sampling phase 2: combine + state machine (lanes 0-15 of wave 0) --------
    if (wave == 0 && lane < ROWS) {
      const int rr = lane;
      if (st_act[rr]) {
        int p = st_pos[rr];
        p = p < 0 ? 0 : (p > TT - 1 ? TT - 1 : p);
        const int choice = st_choice[rr];
        const bool is_const = (choice == CONST_ID);
        st_lp[rr] += st_lpch[rr] + (is_const ? st_lpc[rr] : 0.0f);
        const size_t grow = (size_t)(row0 + rr) * TT + p;
        seq_out[grow] = choice;
        if (is_const) cval_out[grow] = -10.0f + 0.5f * (float)st_cch[rr];
        #pragma unroll
        for (int b = 0; b < BINC; ++b)
          z[rr][BINC * p + b] = (__bf16)(float)((choice >> b) & 1);

        // preorder next position: binary ops push right child & descend left; leaves pop
        const int arity = (choice < 4) ? 2 : ((choice < 6) ? 1 : 0);
        int sp = st_sp[rr];
        if (arity == 2) {
          int spc = sp < 0 ? 0 : (sp > TT - 1 ? TT - 1 : sp);
          st_stack[rr][spc] = 2 * p + 2;
        }
        int sp1 = sp + (arity == 2 ? 1 : 0);
        int tc = sp1 - 1; tc = tc < 0 ? 0 : (tc > TT - 1 ? TT - 1 : tc);
        int topv = st_stack[rr][tc];
        int nxt = (arity != 0) ? (2 * p + 1) : ((sp1 > 0) ? topv : -1);
        int sp2 = (arity == 0) ? ((sp1 - 1 < 0) ? 0 : sp1 - 1) : sp1;
        if (nxt > TT - 1) nxt = -1;
        st_pos[rr] = nxt;
        st_sp[rr]  = sp2;
        st_act[rr] = (nxt >= 0) ? 1 : 0;
      }
    }
  }

  __syncthreads();
  if (wave == 0 && lane < ROWS) lp_out[row0 + lane] = st_lp[lane];
}

extern "C" void kernel_launch(void* const* d_in, const int* in_sizes, int n_in,
                              void* d_out, int out_size, void* d_ws, size_t ws_size,
                              hipStream_t stream) {
  (void)n_in; (void)out_size; (void)d_ws; (void)ws_size;
  const float* x    = (const float*)d_in[0];
  const float* ntok = (const float*)d_in[1];
  const float* ncon = (const float*)d_in[2];
  const float* Wih  = (const float*)d_in[3];
  const float* Whh  = (const float*)d_in[4];
  const float* bih  = (const float*)d_in[5];
  const float* bhh  = (const float*)d_in[6];
  const float* Wout = (const float*)d_in[7];
  const float* bout = (const float*)d_in[8];
  const float* Wc   = (const float*)d_in[9];
  const float* bc   = (const float*)d_in[10];

  const int BS = in_sizes[0] / FEA;

  int*   seq  = (int*)d_out;                              // (BS, T) int32 bits
  float* cval = (float*)d_out + (size_t)BS * TT;          // (BS, T) f32
  float* lp   = (float*)d_out + 2 * (size_t)BS * TT;      // (BS,)   f32

  dim3 grid((unsigned)(BS / ROWS)), block(256);
  hipLaunchKernelGGL(tree_rnn_wmma, grid, block, 0, stream,
                     x, ntok, ncon, Wih, Whh, bih, bhh, Wout, bout, Wc, bc,
                     seq, cval, lp, BS);
}
